// MambaForComposite_31842887533178
// MI455X (gfx1250) — compile-verified, matrix operand order checked by reading, not run on hardware
//
#include <hip/hip_runtime.h>
#include <hip/hip_bf16.h>
#include <math.h>
#include <stdint.h>

#define N_LAYERS  4
#define D_MODEL   256
#define D_STATE   64
#define D_CONV    4
#define D_INNER   512
#define CONV_DIM  640     // D_INNER + 2*D_STATE
#define D_IN_PROJ 1153    // 2*D_INNER + 2*D_STATE + NHEADS
#define VOCAB     100
#define BATCH     8
#define SEQ       2048
#define ROWS      (BATCH*SEQ)   // 16384

typedef __attribute__((ext_vector_type(16))) _Float16 v16h;
typedef __attribute__((ext_vector_type(8)))  float    v8f;
typedef __attribute__((ext_vector_type(4)))  unsigned v4u;
typedef __attribute__((ext_vector_type(8)))  int      v8i;
typedef __attribute__((ext_vector_type(4)))  int      v4i;

__device__ __forceinline__ float silu_f(float x) { return x / (1.f + __expf(-x)); }

// ---------------------------------------------------------------------------
// Embedding gather: hidden[r, d] = embed[ids[r], d]
// ---------------------------------------------------------------------------
__global__ void embed_kernel(const int* __restrict__ ids,
                             const float* __restrict__ emb,
                             float* __restrict__ hid) {
  const int r = blockIdx.x, d = threadIdx.x;
  hid[(size_t)r * D_MODEL + d] = emb[(size_t)ids[r] * D_MODEL + d];
}

// ---------------------------------------------------------------------------
// f32 -> f16 conversion (weights, one shot)
// ---------------------------------------------------------------------------
__global__ void f32_to_f16_kernel(const float* __restrict__ src,
                                  _Float16* __restrict__ dst, size_t n) {
  size_t i = (size_t)blockIdx.x * blockDim.x + threadIdx.x;
  if (i < n) dst[i] = (_Float16)src[i];
}

// ---------------------------------------------------------------------------
// residual = hidden (+ residual); out16 = rmsnorm(residual, w)  (f16)
// ---------------------------------------------------------------------------
__global__ void residual_rmsnorm_kernel(const float* __restrict__ hid,
                                        float* __restrict__ res,
                                        const float* __restrict__ w,
                                        _Float16* __restrict__ out16,
                                        int first) {
  const int r = blockIdx.x, d = threadIdx.x;
  __shared__ float red[D_MODEL];
  float v = hid[(size_t)r * D_MODEL + d];
  if (!first) v += res[(size_t)r * D_MODEL + d];
  res[(size_t)r * D_MODEL + d] = v;
  red[d] = v * v;
  __syncthreads();
  for (int s = D_MODEL / 2; s > 0; s >>= 1) {
    if (d < s) red[d] += red[d + s];
    __syncthreads();
  }
  const float scale = rsqrtf(red[0] * (1.f / D_MODEL) + 1e-5f);
  out16[(size_t)r * D_MODEL + d] = (_Float16)(v * scale * w[d]);
}

// ---------------------------------------------------------------------------
// WMMA f16 GEMM:  C[M x N] = A[M x K] * B[N x K]^T  (f16 in, f32 out)
//
// Block = 128 threads = 4 waves; block tile = 16 (M) x 64 (N).
// The 64 x K f16 weight tile is DMA'd into LDS by the Tensor Data Mover
// (wave 0 issues tensor_load_to_lds, s_wait_tensorcnt 0, then s_barrier).
// D# per ISA ch.8: 2D tensor, data_size=2B, tile 64 x K, tensor_dim1 =
// remaining rows so HW zero-fills the N edge (no divergent bounds code).
// Fragments per ISA 7.12.2: each half-wave's 16 halves are two contiguous
// 16B runs -> 2x global_load_b128 (A) and 2x ds_load_b128 (B) per K-slab.
// ---------------------------------------------------------------------------
template <int K>
__global__ __launch_bounds__(128)
void wmma_gemm_f16(const _Float16* __restrict__ A,
                   const _Float16* __restrict__ B,
                   float* __restrict__ C,
                   int N, int ldc) {
  extern __shared__ _Float16 sB[];      // 64 x K, row-major
  const int nBase = blockIdx.x * 64;
  const int tileM = blockIdx.y;
  const int wave  = threadIdx.x >> 5;
  const int lane  = threadIdx.x & 31;
  const int half  = lane >> 4;
  const int l16   = lane & 15;
  const int rowA  = tileM * 16 + l16;

#if __has_builtin(__builtin_amdgcn_tensor_load_to_lds)
  if (threadIdx.x < 32) {               // one wave issues the TDM op
    const unsigned long long ga =
        (unsigned long long)(uintptr_t)(const void*)(B + (size_t)nBase * K);
    const unsigned lds = (unsigned)(uintptr_t)(void*)sB;  // low 32b = LDS offset
    const unsigned rows = (unsigned)(N - nBase);          // OOB rows -> zero
    v4u g0 = { 1u,                                  // count=1, user mode
               lds,                                 // lds_addr
               (unsigned)ga,                        // global_addr[31:0]
               (unsigned)((ga >> 32) & 0x01FFFFFFu) | (2u << 30) }; // addr hi | type=2
    v8i g1 = { (int)(1u << 16),                     // data_size = 2 bytes
               (int)((unsigned)K << 16),            // tensor_dim0[15:0] @ 63:48
               (int)((rows & 0xFFFFu) << 16),       // tensor_dim0 hi=0 | tensor_dim1 lo
               (int)((unsigned)K << 16),            // tensor_dim1 hi=0 | tile_dim0=K
               64,                                  // tile_dim1=64, tile_dim2=0
               K,                                   // tensor_dim0_stride[31:0]
               0, 0 };                              // stride hi / dim1_stride
    v4i g2 = { 0, 0, 0, 0 };
    v4i g3 = { 0, 0, 0, 0 };
#if defined(__clang_major__) && __clang_major__ >= 23
    v8i g4 = { 0, 0, 0, 0, 0, 0, 0, 0 };
    __builtin_amdgcn_tensor_load_to_lds(g0, g1, g2, g3, g4, 0);
#else
    __builtin_amdgcn_tensor_load_to_lds(g0, g1, g2, g3, 0);
#endif
#if __has_builtin(__builtin_amdgcn_s_wait_tensorcnt)
    __builtin_amdgcn_s_wait_tensorcnt(0);
#endif
  }
  __syncthreads();
#else
  // Fallback: cooperative global->LDS staging with b128 copies
  for (int idx = threadIdx.x; idx < 64 * (K / 8); idx += 128) {
    const int row = idx / (K / 8);
    const int kc  = (idx % (K / 8)) * 8;
    uint4 val = {0u, 0u, 0u, 0u};
    if (nBase + row < N)
      val = *reinterpret_cast<const uint4*>(&B[(size_t)(nBase + row) * K + kc]);
    *reinterpret_cast<uint4*>(&sB[row * K + kc]) = val;
  }
  __syncthreads();
#endif

  const int bRow = wave * 16 + l16;     // B-tile row in LDS == C column
  v8f acc = {};
#pragma unroll
  for (int k0 = 0; k0 < K; k0 += 32) {
    if (k0 + 32 < K)
      __builtin_prefetch(&A[(size_t)rowA * K + k0 + 32], 0, 1);
    union { v16h v; uint4 q[2]; } fa, fb;
    fa.q[0] = *reinterpret_cast<const uint4*>(&A[(size_t)rowA * K + k0 + half * 8]);
    fa.q[1] = *reinterpret_cast<const uint4*>(&A[(size_t)rowA * K + k0 + half * 8 + 16]);
    fb.q[0] = *reinterpret_cast<const uint4*>(&sB[bRow * K + k0 + half * 8]);
    fb.q[1] = *reinterpret_cast<const uint4*>(&sB[bRow * K + k0 + half * 8 + 16]);
    acc = __builtin_amdgcn_wmma_f32_16x16x32_f16(false, fa.v, false, fb.v,
                                                 (short)0, acc, false, false);
  }

  const int col = nBase + wave * 16 + l16;
  if (col < N) {
    const int crow0 = tileM * 16 + (half << 3);
#pragma unroll
    for (int r = 0; r < 8; ++r)
      C[(size_t)(crow0 + r) * ldc + col] = acc[r];
  }
}

// ---------------------------------------------------------------------------
// Depthwise causal conv (k=4) over seq + SiLU; softplus(dt).
// One block per (b,t) row, 640 threads (one per conv channel).
// ---------------------------------------------------------------------------
__global__ void conv_silu_dt_kernel(const float* __restrict__ zxbcdt, // ROWS x 1153
                                    const float* __restrict__ cw,     // CONV_DIM x 4
                                    const float* __restrict__ cb,     // CONV_DIM
                                    const float* __restrict__ dtb,    // scalar
                                    float* __restrict__ xBC,          // ROWS x 640
                                    float* __restrict__ dt) {         // ROWS
  const int r = blockIdx.x;
  const int c = threadIdx.x;
  const int b = r / SEQ, t = r % SEQ;
  float acc = cb[c];
#pragma unroll
  for (int k = 0; k < D_CONV; ++k) {
    const int ts = t - (D_CONV - 1) + k;
    float v = 0.f;
    if (ts >= 0) v = zxbcdt[((size_t)b * SEQ + ts) * D_IN_PROJ + D_INNER + c];
    acc = fmaf(v, cw[c * D_CONV + k], acc);
  }
  xBC[(size_t)r * CONV_DIM + c] = silu_f(acc);
  if (c == 0) {
    float x = zxbcdt[(size_t)r * D_IN_PROJ + (D_IN_PROJ - 1)] + dtb[0];
    dt[r] = (x > 20.f) ? x : log1pf(__expf(x));
  }
}

// ---------------------------------------------------------------------------
// Sequential SSM scan. One block per batch; thread p holds h[p, 0..63] in
// VGPRs; per-step B/C (64+64 f32) broadcast through LDS.
// ---------------------------------------------------------------------------
__global__ void ssm_scan_kernel(const float* __restrict__ xBC, // ROWS x 640
                                const float* __restrict__ dt,  // ROWS
                                const float* __restrict__ alog,
                                const float* __restrict__ dpar,
                                float* __restrict__ y,         // ROWS x 512
                                int layer) {
  const int b = blockIdx.x;
  const int p = threadIdx.x;          // 0..511
  const float Aneg = -__expf(alog[layer]);
  const float Dv   = dpar[layer];
  __shared__ float sB[D_STATE], sC[D_STATE];
  float h[D_STATE];
#pragma unroll
  for (int n = 0; n < D_STATE; ++n) h[n] = 0.f;

  for (int t = 0; t < SEQ; ++t) {
    const size_t row = (size_t)b * SEQ + t;
    const float dtv   = dt[row];
    const float decay = __expf(dtv * Aneg);
    const float xv    = xBC[row * CONV_DIM + p];
    const float dtx   = dtv * xv;
    if (p < D_STATE)          sB[p]           = xBC[row * CONV_DIM + D_INNER + p];
    else if (p < 2 * D_STATE) sC[p - D_STATE] = xBC[row * CONV_DIM + D_INNER + p];
    __syncthreads();
    float accv = 0.f;
#pragma unroll
    for (int n = 0; n < D_STATE; ++n) {
      h[n] = fmaf(h[n], decay, dtx * sB[n]);
      accv = fmaf(h[n], sC[n], accv);
    }
    y[row * D_INNER + p] = fmaf(Dv, xv, accv);
    __syncthreads();
  }
}

// ---------------------------------------------------------------------------
// out16 = rmsnorm(y * silu(z), g)  as f16.  One block per row, 512 threads.
// ---------------------------------------------------------------------------
__global__ void gated_rmsnorm_kernel(const float* __restrict__ y,
                                     const float* __restrict__ zxbcdt,
                                     const float* __restrict__ g,
                                     _Float16* __restrict__ out16) {
  const int r = blockIdx.x, i = threadIdx.x;
  __shared__ float red[D_INNER];
  const float z = zxbcdt[(size_t)r * D_IN_PROJ + i];
  float v = y[(size_t)r * D_INNER + i] * silu_f(z);
  red[i] = v * v;
  __syncthreads();
  for (int s = D_INNER / 2; s > 0; s >>= 1) {
    if (i < s) red[i] += red[i + s];
    __syncthreads();
  }
  const float scale = rsqrtf(red[0] * (1.f / D_INNER) + 1e-5f);
  out16[(size_t)r * D_INNER + i] = (_Float16)(v * scale * g[i]);
}

// ---------------------------------------------------------------------------
// Final: residual + hidden at last token, rmsnorm, logits = fin @ embed^T
// ---------------------------------------------------------------------------
__global__ void final_logits_kernel(const float* __restrict__ hid,
                                    const float* __restrict__ res,
                                    const float* __restrict__ nfw,
                                    const float* __restrict__ emb,
                                    float* __restrict__ out) {
  const int b = blockIdx.x, d = threadIdx.x;
  __shared__ float fin[D_MODEL];
  __shared__ float red[D_MODEL];
  const size_t r = (size_t)b * SEQ + (SEQ - 1);
  float v = hid[r * D_MODEL + d] + res[r * D_MODEL + d];
  red[d] = v * v;
  __syncthreads();
  for (int s = D_MODEL / 2; s > 0; s >>= 1) {
    if (d < s) red[d] += red[d + s];
    __syncthreads();
  }
  const float scale = rsqrtf(red[0] * (1.f / D_MODEL) + 1e-5f);
  fin[d] = v * scale * nfw[d];
  __syncthreads();
  for (int tok = d; tok < VOCAB; tok += D_MODEL) {
    float acc = 0.f;
    for (int k = 0; k < D_MODEL; ++k)
      acc = fmaf(fin[k], emb[(size_t)tok * D_MODEL + k], acc);
    out[b * VOCAB + tok] = acc;
  }
}

// ---------------------------------------------------------------------------
// Host side
// ---------------------------------------------------------------------------
static inline char* ws_align(char* p) {
  return (char*)(((uintptr_t)p + 255) & ~(uintptr_t)255);
}

extern "C" void kernel_launch(void* const* d_in, const int* in_sizes, int n_in,
                              void* d_out, int out_size, void* d_ws, size_t ws_size,
                              hipStream_t stream) {
  (void)in_sizes; (void)n_in; (void)out_size; (void)ws_size;
  const int*   ids        = (const int*)  d_in[0];
  const float* embed      = (const float*)d_in[1];
  const float* in_proj_w  = (const float*)d_in[2];
  const float* conv_w     = (const float*)d_in[3];
  const float* conv_b     = (const float*)d_in[4];
  const float* dt_bias    = (const float*)d_in[5];
  const float* A_log      = (const float*)d_in[6];
  const float* Dparam     = (const float*)d_in[7];
  const float* norm_g_w   = (const float*)d_in[8];
  const float* out_proj_w = (const float*)d_in[9];
  const float* block_nw   = (const float*)d_in[10];
  const float* norm_f_w   = (const float*)d_in[11];
  float* out = (float*)d_out;

  // carve workspace
  char* p = (char*)d_ws;
  float*    residual = (float*)p;    p = ws_align(p + (size_t)ROWS * D_MODEL * 4);
  float*    hidden   = (float*)p;    p = ws_align(p + (size_t)ROWS * D_MODEL * 4);
  _Float16* normed16 = (_Float16*)p; p = ws_align(p + (size_t)ROWS * D_MODEL * 2);
  float*    zxbcdt   = (float*)p;    p = ws_align(p + (size_t)ROWS * D_IN_PROJ * 4);
  float*    xBCc     = (float*)p;    p = ws_align(p + (size_t)ROWS * CONV_DIM * 4);
  float*    dtbuf    = (float*)p;    p = ws_align(p + (size_t)ROWS * 4);
  float*    ybuf     = (float*)p;    p = ws_align(p + (size_t)ROWS * D_INNER * 4);
  _Float16* y16      = (_Float16*)p; p = ws_align(p + (size_t)ROWS * D_INNER * 2);
  _Float16* ipw16    = (_Float16*)p; p = ws_align(p + (size_t)N_LAYERS * D_IN_PROJ * D_MODEL * 2);
  _Float16* opw16    = (_Float16*)p; p = ws_align(p + (size_t)N_LAYERS * D_MODEL * D_INNER * 2);

  embed_kernel<<<ROWS, D_MODEL, 0, stream>>>(ids, embed, hidden);
  {
    size_t nip = (size_t)N_LAYERS * D_IN_PROJ * D_MODEL;
    f32_to_f16_kernel<<<(unsigned)((nip + 255) / 256), 256, 0, stream>>>(in_proj_w, ipw16, nip);
    size_t nop = (size_t)N_LAYERS * D_MODEL * D_INNER;
    f32_to_f16_kernel<<<(unsigned)((nop + 255) / 256), 256, 0, stream>>>(out_proj_w, opw16, nop);
  }

  for (int layer = 0; layer < N_LAYERS; ++layer) {
    residual_rmsnorm_kernel<<<ROWS, D_MODEL, 0, stream>>>(
        hidden, residual, block_nw + (size_t)layer * D_MODEL, normed16, layer == 0);

    {  // in_proj: (ROWS x 256) @ (1153 x 256)^T -> zxbcdt (ROWS x 1153)
      dim3 grid((D_IN_PROJ + 63) / 64, ROWS / 16);
      wmma_gemm_f16<D_MODEL><<<grid, 128, 64 * D_MODEL * 2, stream>>>(
          normed16, ipw16 + (size_t)layer * D_IN_PROJ * D_MODEL,
          zxbcdt, D_IN_PROJ, D_IN_PROJ);
    }

    conv_silu_dt_kernel<<<ROWS, CONV_DIM, 0, stream>>>(
        zxbcdt, conv_w + (size_t)layer * CONV_DIM * D_CONV,
        conv_b + (size_t)layer * CONV_DIM, dt_bias + layer, xBCc, dtbuf);

    ssm_scan_kernel<<<BATCH, D_INNER, 0, stream>>>(
        xBCc, dtbuf, A_log, Dparam, ybuf, layer);

    gated_rmsnorm_kernel<<<ROWS, D_INNER, 0, stream>>>(
        ybuf, zxbcdt, norm_g_w + (size_t)layer * D_INNER, y16);

    {  // out_proj: (ROWS x 512) @ (256 x 512)^T -> hidden (ROWS x 256)
      dim3 grid(D_MODEL / 64, ROWS / 16);
      wmma_gemm_f16<D_INNER><<<grid, 128, 64 * D_INNER * 2, stream>>>(
          y16, opw16 + (size_t)layer * D_MODEL * D_INNER,
          hidden, D_MODEL, D_MODEL);
    }
  }

  final_logits_kernel<<<BATCH, D_MODEL, 0, stream>>>(
      hidden, residual, norm_f_w, embed, out);
}
